// Alignment_60138132079139
// MI455X (gfx1250) — matrix operand order repaired and checked by reading, not compiled, not run on hardware
//
#include <hip/hip_runtime.h>

// Problem constants (from reference): B=32, La=Lb=H=1024
#define BATCH 32
#define LAD   1024
#define LBD   1024
#define HDIM  1024

typedef __attribute__((ext_vector_type(16))) __bf16 v16bf;
typedef __attribute__((ext_vector_type(8)))  __bf16 v8bf;
typedef __attribute__((ext_vector_type(4)))  __bf16 v4bf;
typedef __attribute__((ext_vector_type(8)))  float  v8f;
typedef __attribute__((ext_vector_type(4)))  int    v4i;

#define LDSS 40  // LDS row stride in bf16 (32 data + 8 pad; 80B rows keep 16B alignment)

// Optional gfx1250 async global->LDS path (guarded; falls back to VGPR staging)
#if defined(__HIP_DEVICE_COMPILE__) && __has_builtin(__builtin_amdgcn_global_load_async_to_lds_b128)
#define HAVE_ASYNC_LDS 1
#endif

#ifdef HAVE_ASYNC_LDS
__device__ inline void async_b128(const __bf16* g, const __bf16* l) {
  // Builtin signature (from compiler diagnostic): (AS1 int4*, AS3 int4*, int, int).
  // generic->AS casts via integer: global generic == AS1 numerically,
  // shared generic low 32 bits == LDS offset (aperture layout).
  __builtin_amdgcn_global_load_async_to_lds_b128(
      (__attribute__((address_space(1))) v4i*)(uintptr_t)g,
      (__attribute__((address_space(3))) v4i*)(uint32_t)(uintptr_t)l,
      0, 0);
}
__device__ inline void wait_async0() {
#if __has_builtin(__builtin_amdgcn_s_wait_asynccnt)
  __builtin_amdgcn_s_wait_asynccnt(0);
#else
  asm volatile("s_wait_asynccnt 0x0" ::: "memory");
#endif
}
#endif

// ---------------------------------------------------------------------------
// WMMA fragment loaders (CDNA5 bf16 16x16x32 operand layouts, wave32)
// ---------------------------------------------------------------------------
__device__ inline v16bf load_a_frag(const __bf16* rowptr, int half) {
  v8bf lo = *(const v8bf*)(rowptr + half * 8);
  v8bf hi = *(const v8bf*)(rowptr + half * 8 + 16);
  v16bf r;
#pragma unroll
  for (int i = 0; i < 8; ++i) { r[i] = lo[i]; r[i + 8] = hi[i]; }
  return r;
}

__device__ inline v16bf load_b_frag(const __bf16* rowptr, int half) {
  v8bf lo = *(const v8bf*)(rowptr + half * 8);
  v8bf hi = *(const v8bf*)(rowptr + half * 8 + 16);
  v16bf r;
#pragma unroll
  for (int i = 0; i < 8; ++i) { r[i] = lo[i]; r[i + 8] = hi[i]; }
  return r;
}

// ---------------------------------------------------------------------------
// Kernel 1: f32 -> bf16 conversion of A and B
// ---------------------------------------------------------------------------
__global__ __launch_bounds__(256) void convert_kernel(
    const float* __restrict__ a, const float* __restrict__ b,
    __bf16* __restrict__ Abf, __bf16* __restrict__ Bbf) {
  size_t i = ((size_t)blockIdx.x * 256 + threadIdx.x) * 4;
  float4 va = *(const float4*)(a + i);
  float4 vb = *(const float4*)(b + i);
  v4bf oa, ob;
  oa[0] = (__bf16)va.x; oa[1] = (__bf16)va.y; oa[2] = (__bf16)va.z; oa[3] = (__bf16)va.w;
  ob[0] = (__bf16)vb.x; ob[1] = (__bf16)vb.y; ob[2] = (__bf16)vb.z; ob[3] = (__bf16)vb.w;
  *(v4bf*)(Abf + i) = oa;
  *(v4bf*)(Bbf + i) = ob;
}

// ---------------------------------------------------------------------------
// Kernel 2: S[b,s,t] = mask ? (A . B^T) * temp : -10000  (bf16 WMMA, f32 out)
// 128x128 block tile, 8 waves, wave = 64x32. Double-buffered LDS, software
// pipelined: one barrier per K-chunk; async global->LDS when available.
// ---------------------------------------------------------------------------
__global__ __launch_bounds__(256) void scores_kernel(
    const __bf16* __restrict__ Abf, const __bf16* __restrict__ Bbf,
    const int* __restrict__ mask_a, const int* __restrict__ mask_b,
    const float* __restrict__ temp_p, float* __restrict__ S) {
  const int b    = blockIdx.z;
  const int t0   = blockIdx.x * 128;
  const int s0   = blockIdx.y * 128;
  const int tid  = threadIdx.x;
  const int lane = tid & 31;
  const int wave = tid >> 5;
  const int wm   = wave & 1;
  const int wn   = wave >> 1;
  const int half = lane >> 4;
  const int l15  = lane & 15;

  __shared__ __align__(16) __bf16 Als[2][128][LDSS];
  __shared__ __align__(16) __bf16 Bls[2][128][LDSS];

  const int row  = tid >> 1;
  const int colh = (tid & 1) * 16;
  const __bf16* gA = Abf + (size_t)b * LAD * HDIM + (size_t)(s0 + row) * HDIM + colh;
  const __bf16* gB = Bbf + (size_t)b * LBD * HDIM + (size_t)(t0 + row) * HDIM + colh;

  v8f zero = {0.f, 0.f, 0.f, 0.f, 0.f, 0.f, 0.f, 0.f};
  v8f acc[4][2];
#pragma unroll
  for (int m = 0; m < 4; ++m)
#pragma unroll
    for (int n = 0; n < 2; ++n) acc[m][n] = zero;

#ifdef HAVE_ASYNC_LDS
  // --- async pipeline: wait(i) -> barrier -> issue(i+1) -> wmma(i) ---
  async_b128(gA, &Als[0][row][colh]);
  async_b128(gA + 8, &Als[0][row][colh + 8]);
  async_b128(gB, &Bls[0][row][colh]);
  async_b128(gB + 8, &Bls[0][row][colh + 8]);
  for (int kc = 0; kc < HDIM / 32; ++kc) {
    const int buf = kc & 1;
    wait_async0();
    __syncthreads();
    if (kc + 1 < HDIM / 32) {
      const __bf16* pa = gA + (kc + 1) * 32;
      const __bf16* pb = gB + (kc + 1) * 32;
      async_b128(pa, &Als[buf ^ 1][row][colh]);
      async_b128(pa + 8, &Als[buf ^ 1][row][colh + 8]);
      async_b128(pb, &Bls[buf ^ 1][row][colh]);
      async_b128(pb + 8, &Bls[buf ^ 1][row][colh + 8]);
    }
    v16bf af[4], bf[2];
#pragma unroll
    for (int mt = 0; mt < 4; ++mt)
      af[mt] = load_a_frag(&Als[buf][wm * 64 + mt * 16 + l15][0], half);
#pragma unroll
    for (int nt = 0; nt < 2; ++nt)
      bf[nt] = load_b_frag(&Bls[buf][wn * 32 + nt * 16 + l15][0], half);
#pragma unroll
    for (int mt = 0; mt < 4; ++mt)
#pragma unroll
      for (int nt = 0; nt < 2; ++nt)
        acc[mt][nt] = __builtin_amdgcn_wmma_f32_16x16x32_bf16(
            false, af[mt], false, bf[nt], (short)0, acc[mt][nt], false, false);
  }
#else
  // --- VGPR-staged pipeline: store(i) -> load(i+1) -> barrier -> wmma(i) ---
  v8bf ra0 = *(const v8bf*)gA, ra1 = *(const v8bf*)(gA + 8);
  v8bf rb0 = *(const v8bf*)gB, rb1 = *(const v8bf*)(gB + 8);
  for (int kc = 0; kc < HDIM / 32; ++kc) {
    const int buf = kc & 1;
    *(v8bf*)&Als[buf][row][colh]     = ra0;
    *(v8bf*)&Als[buf][row][colh + 8] = ra1;
    *(v8bf*)&Bls[buf][row][colh]     = rb0;
    *(v8bf*)&Bls[buf][row][colh + 8] = rb1;
    if (kc + 1 < HDIM / 32) {
      const __bf16* pa = gA + (kc + 1) * 32;
      const __bf16* pb = gB + (kc + 1) * 32;
      ra0 = *(const v8bf*)pa; ra1 = *(const v8bf*)(pa + 8);
      rb0 = *(const v8bf*)pb; rb1 = *(const v8bf*)(pb + 8);
    }
    __syncthreads();
    v16bf af[4], bf[2];
#pragma unroll
    for (int mt = 0; mt < 4; ++mt)
      af[mt] = load_a_frag(&Als[buf][wm * 64 + mt * 16 + l15][0], half);
#pragma unroll
    for (int nt = 0; nt < 2; ++nt)
      bf[nt] = load_b_frag(&Bls[buf][wn * 32 + nt * 16 + l15][0], half);
#pragma unroll
    for (int mt = 0; mt < 4; ++mt)
#pragma unroll
      for (int nt = 0; nt < 2; ++nt)
        acc[mt][nt] = __builtin_amdgcn_wmma_f32_16x16x32_bf16(
            false, af[mt], false, bf[nt], (short)0, acc[mt][nt], false, false);
  }
#endif

  // Epilogue: scale + mask + store
  const float temp = *temp_p;
  const int* ma = mask_a + (size_t)b * LAD;
  const int* mb = mask_b + (size_t)b * LBD;
  float* Sb = S + (size_t)b * LAD * LBD;
  int mbv[2];
#pragma unroll
  for (int nt = 0; nt < 2; ++nt) mbv[nt] = mb[t0 + wn * 32 + nt * 16 + l15];
#pragma unroll
  for (int mt = 0; mt < 4; ++mt) {
#pragma unroll
    for (int r = 0; r < 8; ++r) {
      const int s = s0 + wm * 64 + mt * 16 + half * 8 + r;
      const bool mav = ma[s] != 0;
      const size_t rowoff = (size_t)s * LBD;
#pragma unroll
      for (int nt = 0; nt < 2; ++nt) {
        const int t = t0 + wn * 32 + nt * 16 + l15;
        float v = acc[mt][nt][r] * temp;
        if (!(mav && mbv[nt])) v = -10000.0f;
        Sb[rowoff + t] = v;
      }
    }
  }
}

// ---------------------------------------------------------------------------
// Kernel 3: row stats (softmax over t): per (b,s) max and sum(exp)
// ---------------------------------------------------------------------------
__global__ __launch_bounds__(256) void row_stats_kernel(
    const float* __restrict__ S, float* __restrict__ rmax, float* __restrict__ rsum) {
  const int row  = blockIdx.x * 8 + (threadIdx.x >> 5);
  const int lane = threadIdx.x & 31;
  const float* p = S + (size_t)row * LBD;
  float vals[32];
  float m = -INFINITY;
#pragma unroll
  for (int i = 0; i < 32; ++i) { vals[i] = p[lane + i * 32]; m = fmaxf(m, vals[i]); }
#pragma unroll
  for (int off = 16; off > 0; off >>= 1) m = fmaxf(m, __shfl_xor(m, off, 32));
  float sum = 0.f;
#pragma unroll
  for (int i = 0; i < 32; ++i) sum += __expf(vals[i] - m);
#pragma unroll
  for (int off = 16; off > 0; off >>= 1) sum += __shfl_xor(sum, off, 32);
  if (lane == 0) { rmax[row] = m; rsum[row] = sum; }
}

// ---------------------------------------------------------------------------
// Kernel 4a/4b: column stats (softmax over s), segmented online pass
// ---------------------------------------------------------------------------
__global__ __launch_bounds__(256) void col_stats_partial(
    const float* __restrict__ S, float* __restrict__ pmax, float* __restrict__ psum) {
  const int idx = blockIdx.x * 256 + threadIdx.x;
  const int seg = blockIdx.y;
  const int b = idx >> 10, t = idx & 1023;
  const float* p = S + ((size_t)b * LAD + seg * 128) * LBD + t;
  float m = -INFINITY, sum = 0.f;
  for (int s = 0; s < 128; ++s) {
    float v = p[(size_t)s * LBD];
    float nm = fmaxf(m, v);
    sum = sum * __expf(m - nm) + __expf(v - nm);
    m = nm;
  }
  pmax[(size_t)seg * (BATCH * LBD) + idx] = m;
  psum[(size_t)seg * (BATCH * LBD) + idx] = sum;
}

__global__ __launch_bounds__(256) void col_stats_combine(
    const float* __restrict__ pmax, const float* __restrict__ psum,
    float* __restrict__ cmax, float* __restrict__ csum) {
  const int idx = blockIdx.x * 256 + threadIdx.x;
  float m = -INFINITY;
#pragma unroll
  for (int g = 0; g < 8; ++g) m = fmaxf(m, pmax[(size_t)g * (BATCH * LBD) + idx]);
  float s = 0.f;
#pragma unroll
  for (int g = 0; g < 8; ++g)
    s += psum[(size_t)g * (BATCH * LBD) + idx] *
         __expf(pmax[(size_t)g * (BATCH * LBD) + idx] - m);
  cmax[idx] = m;
  csum[idx] = s;
}

// ---------------------------------------------------------------------------
// Kernels 5/6: out[b,m,h] = sum_k P[m,k] * V[k,h]
//   TRANS=false (feature_a): m=s, k=t, P=exp(S[m,k]-rmax[m])/rsum[m], V=Bbf
//   TRANS=true  (feature_b): m=t, k=s, P=exp(S[k,m]-cmax[m])/csum[m], V=Abf
// Double-buffered LDS, pipelined, loop-invariant softmax stats hoisted.
// ---------------------------------------------------------------------------
template <bool TRANS>
__global__ __launch_bounds__(256) void attn_gemm_kernel(
    const float* __restrict__ S, const float* __restrict__ mmax,
    const float* __restrict__ msum, const __bf16* __restrict__ V,
    float* __restrict__ out) {
  const int b    = blockIdx.z;
  const int h0   = blockIdx.x * 128;
  const int m0   = blockIdx.y * 128;
  const int tid  = threadIdx.x;
  const int lane = tid & 31;
  const int wave = tid >> 5;
  const int wm   = wave & 1;
  const int wn   = wave >> 1;
  const int half = lane >> 4;
  const int l15  = lane & 15;

  __shared__ __align__(16) __bf16 Pls[2][128][LDSS];  // [m][k]
  __shared__ __align__(16) __bf16 Vls[2][128][LDSS];  // [n(h)][k]

  const float*  Sb = S + (size_t)b * LAD * LBD;
  const float*  mx = mmax + (size_t)b * 1024;
  const float*  ms = msum + (size_t)b * 1024;
  const __bf16* Vb = V + (size_t)b * 1024 * HDIM;

  // ---- per-thread staging geometry + loop-invariant stats ----
  const int pm  = tid >> 1;            // !TRANS: m-local
  const int pkh = (tid & 1) * 16;      // !TRANS: k-chunk offset
  const int pk  = tid >> 3;            // TRANS: k-local (s)
  const int pmc = (tid & 7) * 16;      // TRANS: m-chunk (t)
  float rmx = 0.f, rinv = 0.f;
  float cmx[16], cinv[16];
  if (!TRANS) {
    rmx  = mx[m0 + pm];
    rinv = 1.0f / ms[m0 + pm];
  } else {
#pragma unroll
    for (int i = 0; i < 16; ++i) {
      cmx[i]  = mx[m0 + pmc + i];
      cinv[i] = 1.0f / ms[m0 + pmc + i];
    }
  }
  // V staging: thread = (k, nc), transposed into LDS
  const int vk  = tid >> 3;
  const int vnc = (tid & 7) * 16;
  const __bf16* gV = Vb + (size_t)vk * HDIM + h0 + vnc;

  v8f zero = {0.f, 0.f, 0.f, 0.f, 0.f, 0.f, 0.f, 0.f};
  v8f acc[4][2];
#pragma unroll
  for (int m = 0; m < 4; ++m)
#pragma unroll
    for (int n = 0; n < 2; ++n) acc[m][n] = zero;

  // ---- prologue: load chunk 0 into regs ----
  float sreg[16];
  v8bf vr0, vr1;
  {
    const float* sp = TRANS ? (Sb + (size_t)pk * LBD + m0 + pmc)
                            : (Sb + (size_t)(m0 + pm) * LBD + pkh);
#pragma unroll
    for (int i = 0; i < 4; ++i) {
      float4 v = *(const float4*)(sp + i * 4);
      sreg[4 * i + 0] = v.x; sreg[4 * i + 1] = v.y;
      sreg[4 * i + 2] = v.z; sreg[4 * i + 3] = v.w;
    }
    vr0 = *(const v8bf*)gV;
    vr1 = *(const v8bf*)(gV + 8);
  }

  for (int kc = 0; kc < 1024 / 32; ++kc) {
    const int buf = kc & 1;
    // ---- compute P (exp) and store P,V tiles into LDS buf ----
    if (!TRANS) {
#pragma unroll
      for (int i = 0; i < 16; ++i)
        Pls[buf][pm][pkh + i] = (__bf16)(__expf(sreg[i] - rmx) * rinv);
    } else {
#pragma unroll
      for (int i = 0; i < 16; ++i)
        Pls[buf][pmc + i][pk] = (__bf16)(__expf(sreg[i] - cmx[i]) * cinv[i]);
    }
#pragma unroll
    for (int i = 0; i < 8; ++i) {
      Vls[buf][vnc + i][vk]     = vr0[i];
      Vls[buf][vnc + 8 + i][vk] = vr1[i];
    }
    // ---- prefetch chunk kc+1 into regs ----
    if (kc + 1 < 1024 / 32) {
      const int k0n = (kc + 1) * 32;
      const float* sp = TRANS ? (Sb + (size_t)(k0n + pk) * LBD + m0 + pmc)
                              : (Sb + (size_t)(m0 + pm) * LBD + k0n + pkh);
#pragma unroll
      for (int i = 0; i < 4; ++i) {
        float4 v = *(const float4*)(sp + i * 4);
        sreg[4 * i + 0] = v.x; sreg[4 * i + 1] = v.y;
        sreg[4 * i + 2] = v.z; sreg[4 * i + 3] = v.w;
      }
      const __bf16* vp = gV + (size_t)k0n * HDIM;
      vr0 = *(const v8bf*)vp;
      vr1 = *(const v8bf*)(vp + 8);
    }
    __syncthreads();
    // ---- WMMA from LDS buf ----
    v16bf af[4], bf[2];
#pragma unroll
    for (int mt = 0; mt < 4; ++mt)
      af[mt] = load_a_frag(&Pls[buf][wm * 64 + mt * 16 + l15][0], half);
#pragma unroll
    for (int nt = 0; nt < 2; ++nt)
      bf[nt] = load_b_frag(&Vls[buf][wn * 32 + nt * 16 + l15][0], half);
#pragma unroll
    for (int mt = 0; mt < 4; ++mt)
#pragma unroll
      for (int nt = 0; nt < 2; ++nt)
        acc[mt][nt] = __builtin_amdgcn_wmma_f32_16x16x32_bf16(
            false, af[mt], false, bf[nt], (short)0, acc[mt][nt], false, false);
  }

  // ---- epilogue: store f32 output [b][m][h] ----
  float* ob = out + (size_t)b * 1024 * HDIM;
#pragma unroll
  for (int mt = 0; mt < 4; ++mt) {
#pragma unroll
    for (int r = 0; r < 8; ++r) {
      const int m = m0 + wm * 64 + mt * 16 + half * 8 + r;
#pragma unroll
      for (int nt = 0; nt < 2; ++nt) {
        const int h = h0 + wn * 32 + nt * 16 + l15;
        ob[(size_t)m * HDIM + h] = acc[mt][nt][r];
      }
    }
  }
}

// ---------------------------------------------------------------------------
// Launch. Workspace layout (bytes from d_ws):
//   [0,64M) A bf16 | [64M,128M) B bf16 | [128M,256M) S f32
//   [256M,...) rmax/rsum/cmax/csum + 8-segment partial col stats (~2.5MB)
// ---------------------------------------------------------------------------
extern "C" void kernel_launch(void* const* d_in, const int* in_sizes, int n_in,
                              void* d_out, int out_size, void* d_ws, size_t ws_size,
                              hipStream_t stream) {
  (void)in_sizes; (void)n_in; (void)out_size; (void)ws_size;
  const float* a      = (const float*)d_in[0];
  const float* b      = (const float*)d_in[1];
  const int*   mask_a = (const int*)d_in[2];
  const int*   mask_b = (const int*)d_in[3];
  const float* temp   = (const float*)d_in[4];
  float* out = (float*)d_out;

  char* ws = (char*)d_ws;
  __bf16* Abf = (__bf16*)(ws);
  __bf16* Bbf = (__bf16*)(ws + ((size_t)64 << 20));
  float*  S   = (float*)(ws + ((size_t)128 << 20));
  float*  rmax = (float*)(ws + ((size_t)256 << 20));
  float*  rsum  = rmax + BATCH * LAD;
  float*  cmax  = rsum + BATCH * LAD;
  float*  csum  = cmax + BATCH * LBD;
  float*  pcmax = csum + BATCH * LBD;
  float*  pcsum = pcmax + 8 * BATCH * LBD;

  convert_kernel<<<(BATCH * LAD * HDIM) / (256 * 4), 256, 0, stream>>>(a, b, Abf, Bbf);

  dim3 g2(LBD / 128, LAD / 128, BATCH);
  scores_kernel<<<g2, 256, 0, stream>>>(Abf, Bbf, mask_a, mask_b, temp, S);

  row_stats_kernel<<<(BATCH * LAD) / 8, 256, 0, stream>>>(S, rmax, rsum);
  col_stats_partial<<<dim3((BATCH * LBD) / 256, 8), 256, 0, stream>>>(S, pcmax, pcsum);
  col_stats_combine<<<(BATCH * LBD) / 256, 256, 0, stream>>>(pcmax, pcsum, cmax, csum);

  dim3 g5(HDIM / 128, LAD / 128, BATCH);
  attn_gemm_kernel<false><<<g5, 256, 0, stream>>>(S, rmax, rsum, Bbf, out);
  attn_gemm_kernel<true><<<g5, 256, 0, stream>>>(S, cmax, csum, Abf,
                                                 out + (size_t)BATCH * LBD * HDIM);
}
